// Speller_1082331759240
// MI455X (gfx1250) — matrix operand order repaired and checked by reading, not compiled, not run on hardware
//
#include <hip/hip_runtime.h>

// ---------------- problem constants ----------------
#define HID    1024
#define BATCH  64
#define SEQ    512
#define LAYERS 4

// ---------------- kernel config --------------------
#define NWG    64      // persistent workgroups; each owns 16 hidden columns
#define NTHR   256     // 8 wave32s
#define JCOLS  16      // hidden columns per WG
#define KP     64      // K-panel width staged in LDS
#define KTOT   (2 * HID)
#define LPAD   16      // LDS row padding (halves) -> 160B row stride, 16B aligned

typedef __attribute__((ext_vector_type(16))) __bf16 v16bf;
typedef __attribute__((ext_vector_type(8)))  float  v8f;
typedef __attribute__((ext_vector_type(4)))  int    v4i;

typedef __attribute__((address_space(1))) v4i gv4i;   // global 16B vector
typedef __attribute__((address_space(3))) v4i lv4i;   // LDS    16B vector

// ---------------- CDNA5 async global->LDS copy ----------------
#if __has_builtin(__builtin_amdgcn_global_load_async_to_lds_b128)
#define HAVE_ASYNC 1
#else
#define HAVE_ASYNC 0
#endif

__device__ __forceinline__ void cp16_g2l(void* lds_dst, const void* gsrc) {
#if HAVE_ASYNC
    __builtin_amdgcn_global_load_async_to_lds_b128(
        (gv4i*)gsrc, (lv4i*)lds_dst, 0, 0);
#else
    *(uint4*)lds_dst = *(const uint4*)gsrc;
#endif
}

__device__ __forceinline__ void async_join() {
#if HAVE_ASYNC
#if __has_builtin(__builtin_amdgcn_s_wait_asynccnt)
    __builtin_amdgcn_s_wait_asynccnt(0);
#else
    asm volatile("s_wait_asynccnt 0x0" ::: "memory");
#endif
#endif
}

// ---------------- small helpers ----------------
__global__ void lstm_init_counter(unsigned* cnt) {
    if (threadIdx.x == 0 && blockIdx.x == 0) *cnt = 0u;
}

__device__ __forceinline__ float sigmoid_f(float x) {
    return 1.0f / (1.0f + __expf(-x));
}

__device__ __forceinline__ float tanh_f(float x) {
    float ax = __builtin_fabsf(x);
    float e  = __expf(-2.0f * ax);
    float r  = (1.0f - e) / (1.0f + e);
    return __builtin_copysignf(r, x);
}

// Device-wide barrier: monotonically increasing counter, one arrival per WG.
__device__ __forceinline__ void grid_barrier(unsigned* cnt, unsigned epoch) {
    __syncthreads();
    if (threadIdx.x == 0) {
        __threadfence();
        __hip_atomic_fetch_add(cnt, 1u, __ATOMIC_RELEASE, __HIP_MEMORY_SCOPE_AGENT);
        const unsigned target = epoch * NWG;
        while (__hip_atomic_load(cnt, __ATOMIC_ACQUIRE, __HIP_MEMORY_SCOPE_AGENT) < target) {
            __builtin_amdgcn_s_sleep(2);
        }
        __threadfence();
    }
    __syncthreads();
}

// ---------------- weight pre-pack (fp32 -> bf16, per-WG gathered rows) ----
// wpk[((l*NWG+wg)*64 + n)*KTOT + k], n = gate*16 + jj,
// k < HID -> W_ih[l][gate*HID + wg*16 + jj][k], else W_hh[...][k-HID]
__global__ void lstm_pack_weights(const float* __restrict__ W_ih,
                                  const float* __restrict__ W_hh,
                                  __bf16* __restrict__ wpk) {
    const size_t nchunk = (size_t)LAYERS * NWG * 64 * (KTOT / 16);
    for (size_t c = (size_t)blockIdx.x * blockDim.x + threadIdx.x;
         c < nchunk; c += (size_t)gridDim.x * blockDim.x) {
        const int    k  = (int)(c % (KTOT / 16)) * 16;
        const size_t r  = c / (KTOT / 16);
        const int    n  = (int)(r % 64);
        const size_t lw = r / 64;
        const int    wg = (int)(lw % NWG);
        const int    l  = (int)(lw / NWG);
        const int    row = (n >> 4) * HID + wg * JCOLS + (n & 15);
        const float* src = ((k < HID) ? W_ih : W_hh)
            + ((size_t)l * 4 * HID + row) * HID + (k & (HID - 1));
        __bf16* dst = wpk + r * KTOT + k;
        #pragma unroll
        for (int q = 0; q < 4; ++q) {
            const float4 v = *reinterpret_cast<const float4*>(src + 4 * q);
            dst[4*q + 0] = (__bf16)v.x;
            dst[4*q + 1] = (__bf16)v.y;
            dst[4*q + 2] = (__bf16)v.z;
            dst[4*q + 3] = (__bf16)v.w;
        }
    }
}

// ---------------- persistent LSTM kernel ----------------
template <bool PACKED>
__global__ void __launch_bounds__(NTHR) lstm_persistent(
    const float*  __restrict__ x,      // [B, T, H] fp32
    const float*  __restrict__ W_ih,   // [L, 4H, H] fp32 (fallback path)
    const float*  __restrict__ W_hh,   // [L, 4H, H] fp32 (fallback path)
    const float*  __restrict__ b_ih,   // [L, 4H]
    const float*  __restrict__ b_hh,   // [L, 4H]
    const __bf16* __restrict__ wpk,    // packed bf16 weights (PACKED path)
    __bf16*       __restrict__ hb,     // [L, B, H] bf16 hidden state (ws)
    float*        __restrict__ out,    // [B, H] fp32
    unsigned*     __restrict__ cnt)
{
    __shared__ __bf16 sInp[BATCH][KP + LPAD];       // input panel  [b][k]
    __shared__ __bf16 sWt [64][KP + LPAD];          // weight panel [n][k]
    __shared__ float  sGates[BATCH][64 + 4];        // gate tile    [b][g*16+jj]
    __shared__ float  sC[LAYERS][BATCH][JCOLS];     // persistent cell state

    const int tid  = threadIdx.x;
    const int wg   = blockIdx.x;
    const int j0   = wg * JCOLS;
    const int lane = tid & 31;
    const int wave = tid >> 5;
    const int mt   = wave >> 1;                     // batch tile 0..3
    const int ntb  = (wave & 1) * 2;                // gate-tile base: 0 or 2

    // staging mapping: thread -> row bi (0..63), 16 k's starting at kk0
    const int bi  = tid >> 2;
    const int kk0 = (tid & 3) * 16;

    unsigned epoch = 0;

    // ---- init: zero cell state (LDS) and my h columns (global bf16) ----
    for (int e = tid; e < BATCH * JCOLS; e += NTHR) {
        const int b = e >> 4, jj = e & 15;
        for (int l = 0; l < LAYERS; ++l) {
            sC[l][b][jj] = 0.0f;
            hb[(size_t)l * BATCH * HID + (size_t)b * HID + j0 + jj] = (__bf16)0.0f;
        }
    }
    grid_barrier(cnt, ++epoch);

    for (int t = 0; t < SEQ; ++t) {
        for (int l = 0; l < LAYERS; ++l) {
            v8f acc0 = {0.f,0.f,0.f,0.f,0.f,0.f,0.f,0.f};
            v8f acc1 = {0.f,0.f,0.f,0.f,0.f,0.f,0.f,0.f};

            // k-tile MAC block (A: sInp, B: sWt rows = weight rows)
            auto do_ktiles = [&]() {
                #pragma unroll
                for (int kt = 0; kt < KP; kt += 32) {
                    const int gg = (lane >> 4) & 1;
                    v16bf a;
                    {
                        const int m = mt * 16 + (lane & 15);
                        #pragma unroll
                        for (int v = 0; v < 8; ++v) {
                            const int k = kt + gg * 16 + 2 * v;
                            a[2*v]     = sInp[m][k];
                            a[2*v + 1] = sInp[m][k + 1];
                        }
                    }
                    const int ncol = lane & 15;
                    const int kb   = kt + gg * 16;
                    v16bf b0, b1;
                    #pragma unroll
                    for (int v = 0; v < 8; ++v) {
                        b0[2*v]     = sWt[ntb * 16 + ncol][kb + 2*v];
                        b0[2*v + 1] = sWt[ntb * 16 + ncol][kb + 2*v + 1];
                        b1[2*v]     = sWt[(ntb + 1) * 16 + ncol][kb + 2*v];
                        b1[2*v + 1] = sWt[(ntb + 1) * 16 + ncol][kb + 2*v + 1];
                    }
                    acc0 = __builtin_amdgcn_wmma_f32_16x16x32_bf16(
                        false, a, false, b0, (short)0, acc0, false, false);
                    acc1 = __builtin_amdgcn_wmma_f32_16x16x32_bf16(
                        false, a, false, b1, (short)0, acc1, false, false);
                }
            };

            // fp32 -> bf16 cvt + LDS store of 16 values
            auto stage_cvt = [&](__bf16* dst, const float* src) {
                #pragma unroll
                for (int q = 0; q < 4; ++q) {
                    const float4 v = *reinterpret_cast<const float4*>(src + q * 4);
                    dst[q*4 + 0] = (__bf16)v.x;
                    dst[q*4 + 1] = (__bf16)v.y;
                    dst[q*4 + 2] = (__bf16)v.z;
                    dst[q*4 + 3] = (__bf16)v.w;
                }
            };

            const int  wrow    = (bi >> 4) * HID + j0 + (bi & 15);   // fp32 path
            const size_t wpkrow = (((size_t)l * NWG + wg) * 64 + bi) * KTOT; // bf16 path
            const __bf16* inPrev = hb + (size_t)(l - 1) * BATCH * HID;
            const __bf16* inCur  = hb + (size_t)l * BATCH * HID;

            // ================= X phase: k in [0, HID) =================
            for (int panel = 0; panel < HID / KP; ++panel) {
                const int kbase = panel * KP;
                __syncthreads();
                if (l == 0) {
                    stage_cvt(&sInp[bi][kk0],
                              x + ((size_t)bi * SEQ + t) * HID + kbase + kk0);
                } else {
                    const __bf16* src = inPrev + (size_t)bi * HID + kbase + kk0;
                    cp16_g2l(&sInp[bi][kk0],     src);
                    cp16_g2l(&sInp[bi][kk0 + 8], src + 8);
                }
                if (PACKED) {
                    const __bf16* wsrc = wpk + wpkrow + kbase + kk0;
                    cp16_g2l(&sWt[bi][kk0],     wsrc);
                    cp16_g2l(&sWt[bi][kk0 + 8], wsrc + 8);
                    __builtin_prefetch(wsrc + KP, 0, 1);
                } else {
                    const float* wsrc = W_ih + ((size_t)l * 4 * HID + wrow) * HID
                                             + kbase + kk0;
                    stage_cvt(&sWt[bi][kk0], wsrc);
                    __builtin_prefetch(wsrc + KP, 0, 1);
                }
                async_join();
                __syncthreads();
                do_ktiles();
            }

            // ================= H phase: k in [HID, 2*HID) =================
            for (int panel = 0; panel < HID / KP; ++panel) {
                const int kbase = panel * KP;
                __syncthreads();
                {
                    const __bf16* src = inCur + (size_t)bi * HID + kbase + kk0;
                    cp16_g2l(&sInp[bi][kk0],     src);
                    cp16_g2l(&sInp[bi][kk0 + 8], src + 8);
                }
                if (PACKED) {
                    const __bf16* wsrc = wpk + wpkrow + HID + kbase + kk0;
                    cp16_g2l(&sWt[bi][kk0],     wsrc);
                    cp16_g2l(&sWt[bi][kk0 + 8], wsrc + 8);
                    __builtin_prefetch(wsrc + KP, 0, 1);
                } else {
                    const float* wsrc = W_hh + ((size_t)l * 4 * HID + wrow) * HID
                                             + kbase + kk0;
                    stage_cvt(&sWt[bi][kk0], wsrc);
                    __builtin_prefetch(wsrc + KP, 0, 1);
                }
                async_join();
                __syncthreads();
                do_ktiles();
            }

            // ---- epilogue: scatter C/D fragments into sGates ----
            __syncthreads();
            {
                const int gg = (lane >> 4) & 1;
                const int nn = lane & 15;
                #pragma unroll
                for (int r = 0; r < 8; ++r) {
                    const int mm = mt * 16 + gg * 8 + r;
                    sGates[mm][ntb * 16 + nn]       = acc0[r];
                    sGates[mm][(ntb + 1) * 16 + nn] = acc1[r];
                }
            }
            grid_barrier(cnt, ++epoch);   // all reads of h(t-1) complete

            // ---- elementwise LSTM cell update for owned columns ----
            #pragma unroll
            for (int q = 0; q < (BATCH * JCOLS) / NTHR; ++q) {
                const int e  = tid + q * NTHR;
                const int b  = e >> 4, jj = e & 15;
                const int nb = l * 4 * HID + j0 + jj;
                const float ig = sGates[b][jj]      + b_ih[nb]          + b_hh[nb];
                const float fg = sGates[b][16 + jj] + b_ih[nb + HID]    + b_hh[nb + HID];
                const float gv = sGates[b][32 + jj] + b_ih[nb + 2*HID]  + b_hh[nb + 2*HID];
                const float og = sGates[b][48 + jj] + b_ih[nb + 3*HID]  + b_hh[nb + 3*HID];
                const float c_old = sC[l][b][jj];
                const float c_new = sigmoid_f(fg) * c_old + sigmoid_f(ig) * tanh_f(gv);
                const float h_new = sigmoid_f(og) * tanh_f(c_new);
                sC[l][b][jj] = c_new;
                hb[(size_t)l * BATCH * HID + (size_t)b * HID + j0 + jj] = (__bf16)h_new;
                if (t == SEQ - 1 && l == LAYERS - 1)
                    out[(size_t)b * HID + j0 + jj] = h_new;
            }
            grid_barrier(cnt, ++epoch);   // h(t) visible to everyone
        }
    }
}

extern "C" void kernel_launch(void* const* d_in, const int* in_sizes, int n_in,
                              void* d_out, int out_size, void* d_ws, size_t ws_size,
                              hipStream_t stream) {
    const float* x    = (const float*)d_in[0];
    const float* W_ih = (const float*)d_in[1];
    const float* W_hh = (const float*)d_in[2];
    const float* b_ih = (const float*)d_in[3];
    const float* b_hh = (const float*)d_in[4];
    float* out = (float*)d_out;

    // workspace layout
    unsigned* cnt = (unsigned*)d_ws;
    __bf16*   hb  = (__bf16*)((char*)d_ws + 1024);          // 512 KB
    const size_t off_wpk   = (size_t)1 << 20;               // 1 MB
    const size_t wpk_bytes = (size_t)LAYERS * NWG * 64 * KTOT * 2;  // 64 MB
    __bf16*   wpk = (__bf16*)((char*)d_ws + off_wpk);
    const bool packed = (ws_size >= off_wpk + wpk_bytes);

    lstm_init_counter<<<1, 32, 0, stream>>>(cnt);
    if (packed) {
        lstm_pack_weights<<<2048, 256, 0, stream>>>(W_ih, W_hh, wpk);
        lstm_persistent<true><<<NWG, NTHR, 0, stream>>>(
            x, W_ih, W_hh, b_ih, b_hh, wpk, hb, out, cnt);
    } else {
        lstm_persistent<false><<<NWG, NTHR, 0, stream>>>(
            x, W_ih, W_hh, b_ih, b_hh, wpk, hb, out, cnt);
    }
}
